// GNNLotteryModel_62105227100528
// MI455X (gfx1250) — matrix operand order repaired and checked by reading, not compiled, not run on hardware
//
#include <hip/hip_runtime.h>
#include <math.h>

typedef __bf16 bf16;
typedef __attribute__((ext_vector_type(16))) __bf16 v16bf;
typedef __attribute__((ext_vector_type(8)))  __bf16 v8bf;
typedef __attribute__((ext_vector_type(8)))  float  v8f;

#define HEADS 8
#define CH1 512   /* H*C1 */
#define CH2 256   /* H*C2 */
#define C3V 16
#define NEG_SLOPE 0.2f

__device__ __forceinline__ float lrelu(float x){ return x > 0.f ? x : NEG_SLOPE * x; }
__device__ __forceinline__ float elu1(float x){ return x > 0.f ? x : (expf(x) - 1.f); }
__device__ __forceinline__ unsigned fkey(float f){
  unsigned u = __float_as_uint(f);
  return (u & 0x80000000u) ? ~u : (u | 0x80000000u);
}
__device__ __forceinline__ float funkey(unsigned k){
  unsigned u = (k & 0x80000000u) ? (k & 0x7fffffffu) : ~k;
  return __uint_as_float(u);
}

// ---------------- fills ----------------
__global__ void k_fill_f32(float* p, float v, long n){
  long i = (long)blockIdx.x * blockDim.x + threadIdx.x;
  if (i < n) p[i] = v;
}
__global__ void k_fill_u32(unsigned* p, unsigned v, long n){
  long i = (long)blockIdx.x * blockDim.x + threadIdx.x;
  if (i < n) p[i] = v;
}

// ---------------- layer 1 linear: (N,5)@(5,512) -> bf16 ----------------
__global__ void k_lin1(const float* __restrict__ x, const float* __restrict__ W,
                       bf16* __restrict__ h, int N){
  long t = (long)blockIdx.x * blockDim.x + threadIdx.x;
  if (t >= (long)N * CH1) return;
  int c = (int)(t & (CH1 - 1));
  long n = t >> 9;
  const float* xr = x + n * 5;
  float acc = 0.f;
  #pragma unroll
  for (int k = 0; k < 5; ++k) acc += xr[k] * W[k * CH1 + c];
  h[t] = (bf16)acc;
}

// ---------------- per-head attention scores ----------------
__global__ void k_scores(const bf16* __restrict__ h, const float* __restrict__ a_s,
                         const float* __restrict__ a_d, float* __restrict__ es,
                         float* __restrict__ ed, int N, int C){
  int t = blockIdx.x * blockDim.x + threadIdx.x;
  if (t >= N * HEADS) return;
  int hh = t & (HEADS - 1);
  long n = t >> 3;
  const bf16* hp = h + ((long)n * HEADS + hh) * C;
  const float* ap = a_s + hh * C;
  const float* bp = a_d + hh * C;
  float ss = 0.f, dd = 0.f;
  for (int c = 0; c < C; ++c){
    float v = (float)hp[c];
    ss += v * ap[c]; dd += v * bp[c];
  }
  es[t] = ss; ed[t] = dd;
}

// ---------------- edge softmax: max / denom / aggregate ----------------
__global__ void k_edge_max(const int* __restrict__ ei, const float* __restrict__ es,
                           const float* __restrict__ ed, unsigned* __restrict__ mkey,
                           int E, int N){
  long t = (long)blockIdx.x * blockDim.x + threadIdx.x;
  long total = (long)(E + N) * HEADS;
  if (t >= total) return;
  int hh = (int)(t & (HEADS - 1));
  int e = (int)(t >> 3);
  int s, d;
  if (e < E){ s = ei[e]; d = ei[E + e]; } else { s = d = e - E; }
  float v = lrelu(es[(long)s * HEADS + hh] + ed[(long)d * HEADS + hh]);
  atomicMax(&mkey[(long)d * HEADS + hh], fkey(v));
}

__global__ void k_decode_m(const unsigned* __restrict__ mk, float* __restrict__ mf, long n){
  long t = (long)blockIdx.x * blockDim.x + threadIdx.x;
  if (t >= n) return;
  float f = funkey(mk[t]);
  if (!isfinite(f)) f = 0.f;
  mf[t] = f;
}

__global__ void k_edge_den(const int* __restrict__ ei, const float* __restrict__ es,
                           const float* __restrict__ ed, const float* __restrict__ m,
                           float* __restrict__ den, int E, int N){
  long t = (long)blockIdx.x * blockDim.x + threadIdx.x;
  long total = (long)(E + N) * HEADS;
  if (t >= total) return;
  int hh = (int)(t & (HEADS - 1));
  int e = (int)(t >> 3);
  int s, d;
  if (e < E){ s = ei[e]; d = ei[E + e]; } else { s = d = e - E; }
  long ih = (long)d * HEADS + hh;
  float v = lrelu(es[(long)s * HEADS + hh] + ed[ih]);
  atomicAdd(&den[ih], expf(v - m[ih]));
}

__global__ void k_edge_agg(const int* __restrict__ ei, const bf16* __restrict__ h,
                           const float* __restrict__ es, const float* __restrict__ ed,
                           const float* __restrict__ m, const float* __restrict__ den,
                           float* __restrict__ out, int E, int N, int C){
  long t = (long)blockIdx.x * blockDim.x + threadIdx.x;
  long total = (long)(E + N) * HEADS * C;
  if (t >= total) return;
  int c = (int)(t % C);
  long eh = t / C;
  int hh = (int)(eh & (HEADS - 1));
  int e = (int)(eh >> 3);
  int s, d;
  if (e < E){ s = ei[e]; d = ei[E + e]; } else { s = d = e - E; }
  long ih = (long)d * HEADS + hh;
  float v = lrelu(es[(long)s * HEADS + hh] + ed[ih]);
  float alpha = expf(v - m[ih]) / (den[ih] + 1e-16f);
  float hv = (float)h[((long)s * HEADS + hh) * C + c];
  atomicAdd(&out[ih * C + c], hv * alpha);
}

// ---------------- bias + ELU (+ optional bf16 cast) ----------------
__global__ void k_bias_elu_bf(const float* __restrict__ in, const float* __restrict__ b,
                              bf16* __restrict__ out, long total, int CH){
  long t = (long)blockIdx.x * blockDim.x + threadIdx.x;
  if (t >= total) return;
  int c = (int)(t % CH);
  out[t] = (bf16)elu1(in[t] + b[c]);
}
__global__ void k_bias_elu_ip(float* __restrict__ p, const float* __restrict__ b,
                              long total, int CH){
  long t = (long)blockIdx.x * blockDim.x + threadIdx.x;
  if (t >= total) return;
  int c = (int)(t % CH);
  p[t] = elu1(p[t] + b[c]);
}

// ---------------- W2 fragment packing for WMMA B-operand ----------------
// B 32x16 (16-bit) lane layout: element j, lanes 0-15: K=j (j<8), K=j+8 (j>=8) ; lanes 16-31: K=8+j (j<8), K=16+j (j>=8).
__global__ void k_pack_w2(const float* __restrict__ W2, bf16* __restrict__ Bp,
                          int K, int Ncols){
  int t = blockIdx.x * blockDim.x + threadIdx.x;
  if (t >= K * Ncols) return;
  int j    = t & 15;
  int lane = (t >> 4) & 31;
  int rem  = t >> 9;
  int NT   = Ncols >> 4;
  int nt   = rem % NT;
  int kt   = rem / NT;
  int k = kt * 32 + ((lane & 16) ? 8 : 0) + (j < 8 ? j : j + 8);
  int n = nt * 16 + (lane & 15);
  Bp[t] = (bf16)W2[k * Ncols + n];
}

// ---------------- WMMA GEMM: C(M,NC) = A(M,K)bf16 @ Bpacked, f32 acc ----------------
// Compile-time K/NC so every fragment address folds into immediate offsets
// (A row stride = K*2 B, B k-step stride = (NC/16)*32*16*2 B).
template<int K, int NC>
__global__ __launch_bounds__(256)
void k_gemm_bf16(const bf16* __restrict__ A, const bf16* __restrict__ Bp,
                 bf16* __restrict__ Cd, int M){
  constexpr int NT = NC >> 4;
  const int lane = threadIdx.x & 31;
  const int wid  = threadIdx.x >> 5;
  int tile = blockIdx.x * (blockDim.x >> 5) + wid;
  int total = (M >> 4) * NT;
  if (tile >= total) return;              // wave-uniform exit, EXEC stays full
  int rt = tile / NT, ct = tile - rt * NT;
  int m0 = rt << 4, n0 = ct << 4;
  const int hi8 = (lane & 16) ? 8 : 0;    // A: lanes>=16 start at K+8 ; C/D: rows +8
  const bf16* __restrict__ arow  = A + ((size_t)(m0 + (lane & 15)) << 9) + hi8; // K=512 -> <<9
  const bf16* __restrict__ bbase = Bp + ((size_t)ct * 32 + lane) * 16;
  v8f acc = {};
  #pragma unroll
  for (int ks = 0; ks < K / 32; ++ks){
    v8bf alo = *(const v8bf*)(arow + ks * 32);        // K = base..base+7
    v8bf ahi = *(const v8bf*)(arow + ks * 32 + 16);   // K = base+16..base+23
    v16bf a = __builtin_shufflevector(alo, ahi, 0,1,2,3,4,5,6,7,8,9,10,11,12,13,14,15);
    v16bf b = *(const v16bf*)(bbase + (size_t)ks * (NT * 512)); // 16 KB immediate stride
    acc = __builtin_amdgcn_wmma_f32_16x16x32_bf16(false, a, false, b,
                                                  (short)0, acc, false, false);
  }
  int cc = n0 + (lane & 15);
  #pragma unroll
  for (int r = 0; r < 8; ++r)
    Cd[(size_t)(m0 + hi8 + r) * NC + cc] = (bf16)acc[r];
}

// ---------------- GCN ----------------
__global__ void k_lin_g(const float* __restrict__ X, const float* __restrict__ W,
                        float* __restrict__ out, int N){
  long t = (long)blockIdx.x * blockDim.x + threadIdx.x;
  if (t >= (long)N * C3V) return;
  int c = (int)(t & (C3V - 1));
  long n = t >> 4;
  const float* xr = X + n * CH2;
  float acc = 0.f;
  for (int k = 0; k < CH2; ++k) acc += xr[k] * W[k * C3V + c];
  out[t] = acc;
}

__global__ void k_deg(const int* __restrict__ ei, const float* __restrict__ ew,
                      float* __restrict__ deg, int E){
  int e = blockIdx.x * blockDim.x + threadIdx.x;
  if (e >= E) return;
  atomicAdd(&deg[ei[E + e]], ew[e]);
}
__global__ void k_dinv(const float* __restrict__ deg, float* __restrict__ dinv, int N){
  int n = blockIdx.x * blockDim.x + threadIdx.x;
  if (n >= N) return;
  float d = deg[n];
  dinv[n] = d > 0.f ? rsqrtf(d) : 0.f;
}
__global__ void k_gcn_agg(const int* __restrict__ ei, const float* __restrict__ ew,
                          const float* __restrict__ h3, const float* __restrict__ dinv,
                          float* __restrict__ out3, int E, int N){
  long t = (long)blockIdx.x * blockDim.x + threadIdx.x;
  long total = (long)(E + N) * C3V;
  if (t >= total) return;
  int c = (int)(t & (C3V - 1));
  int e = (int)(t >> 4);
  int s, d; float wv;
  if (e < E){ s = ei[e]; d = ei[E + e]; wv = ew[e]; } else { s = d = e - E; wv = 1.f; }
  float norm = dinv[s] * wv * dinv[d];
  atomicAdd(&out3[(long)d * C3V + c], h3[(long)s * C3V + c] * norm);
}

// ---------------- head ----------------
__global__ void k_final(const float* __restrict__ o3, const float* __restrict__ bg,
                        const float* __restrict__ Wfc, const float* __restrict__ bfc,
                        float* __restrict__ out, int N){
  int n = blockIdx.x * blockDim.x + threadIdx.x;
  if (n >= N) return;
  float s = 0.f;
  #pragma unroll
  for (int c = 0; c < C3V; ++c) s += elu1(o3[(long)n * C3V + c] + bg[c]) * Wfc[c];
  s += bfc[0];
  out[n] = 1.f / (1.f + expf(-s));
}

extern "C" void kernel_launch(void* const* d_in, const int* in_sizes, int n_in,
                              void* d_out, int out_size, void* d_ws, size_t ws_size,
                              hipStream_t stream) {
  const float* x   = (const float*)d_in[0];
  const int*   ei  = (const int*)  d_in[1];
  const float* ew  = (const float*)d_in[2];
  const float* W1  = (const float*)d_in[3];
  const float* as1 = (const float*)d_in[4];
  const float* ad1 = (const float*)d_in[5];
  const float* b1  = (const float*)d_in[6];
  const float* W2  = (const float*)d_in[7];
  const float* as2 = (const float*)d_in[8];
  const float* ad2 = (const float*)d_in[9];
  const float* b2  = (const float*)d_in[10];
  const float* Wg  = (const float*)d_in[11];
  const float* bg  = (const float*)d_in[12];
  const float* Wfc = (const float*)d_in[13];
  const float* bfc = (const float*)d_in[14];
  float* out = (float*)d_out;

  const int N = in_sizes[0] / 5;
  const int E = in_sizes[1] / 2;
  const long ET = (long)E + N;

  // ---- workspace partition (aliased; ~240 MB) ----
  size_t off = 0;
  auto alloc = [&](size_t bytes) -> void* {
    void* p = (char*)d_ws + off;
    off += (bytes + 255) & ~(size_t)255;
    return p;
  };
  bf16*  h1   = (bf16*)  alloc((size_t)N * CH1 * 2);   // layer-1 features (bf16)
  float* es   = (float*) alloc((size_t)N * HEADS * 4);
  float* ed   = (float*) alloc((size_t)N * HEADS * 4);
  void*  mraw =          alloc((size_t)N * HEADS * 4); // ordered-uint max, then f32
  float* den  = (float*) alloc((size_t)N * HEADS * 4);
  float* agg  = (float*) alloc((size_t)N * CH1 * 4);   // f32 accumulator (L2-resident)
  bf16*  x2   = (bf16*)  alloc((size_t)N * CH1 * 2);   // layer-2 input (bf16, GEMM A)
  bf16*  w2p  = (bf16*)  alloc((size_t)CH1 * CH2 * 2); // packed W2 fragments
  bf16*  h2   = (bf16*)  alloc((size_t)N * CH2 * 2);   // layer-2 features (bf16)
  unsigned* mkey = (unsigned*)mraw;
  float*    mf   = (float*)mraw;
  // GCN scratch aliases the (dead by then) h1 region:
  float* h3   = (float*)h1;
  float* degb = h3 + (size_t)N * C3V;
  float* dinv = degb + N;
  float* out3 = dinv + N;

  const unsigned KEY_NEG_INF = 0x007FFFFFu; // fkey(-inf)
  auto nb = [](long n){ return (unsigned)((n + 255) / 256); };
  #define L(kern, n, ...) kern<<<nb(n), 256, 0, stream>>>(__VA_ARGS__)

  // ================= GAT layer 1 (C=64) =================
  L(k_lin1, (long)N * CH1, x, W1, h1, N);
  L(k_scores, (long)N * HEADS, h1, as1, ad1, es, ed, N, 64);
  L(k_fill_u32, (long)N * HEADS, mkey, KEY_NEG_INF, (long)N * HEADS);
  L(k_fill_f32, (long)N * HEADS, den, 0.f, (long)N * HEADS);
  L(k_fill_f32, (long)N * CH1, agg, 0.f, (long)N * CH1);
  L(k_edge_max, ET * HEADS, ei, es, ed, mkey, E, N);
  L(k_decode_m, (long)N * HEADS, mkey, mf, (long)N * HEADS);
  L(k_edge_den, ET * HEADS, ei, es, ed, mf, den, E, N);
  L(k_edge_agg, ET * CH1, ei, h1, es, ed, mf, den, agg, E, N, 64);
  L(k_bias_elu_bf, (long)N * CH1, agg, b1, x2, (long)N * CH1, CH1);

  // ================= GAT layer 2: WMMA GEMM then attention (C=32) =================
  L(k_pack_w2, (long)CH1 * CH2, W2, w2p, CH1, CH2);
  {
    int tiles = (N >> 4) * (CH2 >> 4);             // 3125 * 16 = 50000
    unsigned blocks = (unsigned)((tiles + 7) / 8); // 8 waves/block
    k_gemm_bf16<CH1, CH2><<<blocks, 256, 0, stream>>>(x2, w2p, h2, N);
  }
  L(k_scores, (long)N * HEADS, h2, as2, ad2, es, ed, N, 32);
  L(k_fill_u32, (long)N * HEADS, mkey, KEY_NEG_INF, (long)N * HEADS);
  L(k_fill_f32, (long)N * HEADS, den, 0.f, (long)N * HEADS);
  L(k_fill_f32, (long)N * CH2, agg, 0.f, (long)N * CH2);
  L(k_edge_max, ET * HEADS, ei, es, ed, mkey, E, N);
  L(k_decode_m, (long)N * HEADS, mkey, mf, (long)N * HEADS);
  L(k_edge_den, ET * HEADS, ei, es, ed, mf, den, E, N);
  L(k_edge_agg, ET * CH2, ei, h2, es, ed, mf, den, agg, E, N, 32);
  L(k_bias_elu_ip, (long)N * CH2, agg, b2, (long)N * CH2, CH2); // agg now = x3 (f32)

  // ================= GCN layer =================
  L(k_lin_g, (long)N * C3V, agg, Wg, h3, N);
  L(k_fill_f32, (long)N, degb, 1.0f, (long)N);  // self-loop weight 1 folded in
  L(k_deg, (long)E, ei, ew, degb, E);
  L(k_dinv, (long)N, degb, dinv, N);
  L(k_fill_f32, (long)N * C3V, out3, 0.f, (long)N * C3V);
  L(k_gcn_agg, ET * C3V, ei, ew, h3, dinv, out3, E, N);

  // ================= head =================
  L(k_final, (long)N, out3, bg, Wfc, bfc, out, N);
  #undef L
}
